// EquiformerPointHead_8521215115335
// MI455X (gfx1250) — matrix operand order repaired and checked by reading, not compiled, not run on hardware
//
#include <hip/hip_runtime.h>
#include <math.h>

typedef _Float16 half_t;
typedef __attribute__((ext_vector_type(16))) _Float16 v16h;
typedef __attribute__((ext_vector_type(8)))  _Float16 v8h;
typedef __attribute__((ext_vector_type(8)))  float    v8f;

#define NPTS 10000
#define DIN  256
#define C0   256
#define C1   64
#define DEPTH 4
#define HEADS 4
#define DH   32
#define HDIM 128
#define KNN  16
#define RBFN 16
#define DOUT 256
#define NK   (NPTS * KNN)

// ---------------------------------------------------------------------------
// f32 -> f16 conversion
// ---------------------------------------------------------------------------
__global__ void cvt_f16_kernel(const float* __restrict__ src, half_t* __restrict__ dst, int n) {
    int i = blockIdx.x * blockDim.x + threadIdx.x;
    if (i < n) dst[i] = (half_t)src[i];
}

// ---------------------------------------------------------------------------
// Brute-force kNN: one block per query, d2 array lives in LDS (40KB of 320KB)
// ---------------------------------------------------------------------------
__global__ void knn_kernel(const float* __restrict__ coors, const unsigned char* __restrict__ mask,
                           int* __restrict__ idx, int n, int k) {
    extern __shared__ float sm[];
    float* d2 = sm;                       // n floats
    float* rv = sm + n;                   // blockDim floats
    int*   ri = (int*)(sm + n + blockDim.x);
    int q = blockIdx.x;
    float qx = coors[q*3+0], qy = coors[q*3+1], qz = coors[q*3+2];
    for (int j = threadIdx.x; j < n; j += blockDim.x) {
        float dx = coors[j*3+0] - qx, dy = coors[j*3+1] - qy, dz = coors[j*3+2] - qz;
        float dd = dx*dx + dy*dy + dz*dz;
        d2[j] = mask[j] ? dd : INFINITY;
    }
    __syncthreads();
    for (int t = 0; t < k; t++) {
        float best = INFINITY; int bi = 0x7fffffff;
        for (int j = threadIdx.x; j < n; j += blockDim.x) {
            float v = d2[j];
            if (v < best || (v == best && j < bi)) { best = v; bi = j; }
        }
        rv[threadIdx.x] = best; ri[threadIdx.x] = bi;
        __syncthreads();
        for (int s = blockDim.x >> 1; s > 0; s >>= 1) {
            if (threadIdx.x < s) {
                float ov = rv[threadIdx.x + s]; int oi = ri[threadIdx.x + s];
                if (ov < rv[threadIdx.x] || (ov == rv[threadIdx.x] && oi < ri[threadIdx.x])) {
                    rv[threadIdx.x] = ov; ri[threadIdx.x] = oi;
                }
            }
            __syncthreads();
        }
        if (threadIdx.x == 0) { idx[q*k + t] = ri[0]; d2[ri[0]] = INFINITY; }
        __syncthreads();
    }
}

// ---------------------------------------------------------------------------
// Geometry: rel_unit, RBF, neighbor mask, gather row maps for vector features
// ---------------------------------------------------------------------------
__global__ void geom_kernel(const float* __restrict__ coors, const unsigned char* __restrict__ mask,
                            const int* __restrict__ idx, float* __restrict__ rel_unit,
                            float* __restrict__ rbf, float* __restrict__ nbm,
                            int* __restrict__ rowmapV, int nn, int kk) {
    int i = blockIdx.x * blockDim.x + threadIdx.x;
    if (i >= nn * kk) return;
    int q = i / kk;
    int j = idx[i];
    float dx = coors[j*3+0] - coors[q*3+0];
    float dy = coors[j*3+1] - coors[q*3+1];
    float dz = coors[j*3+2] - coors[q*3+2];
    float d2 = dx*dx + dy*dy + dz*dz;
    float dist = sqrtf(d2 + 1e-6f);
    float inv = 1.0f / dist;
    rel_unit[i*3+0] = dx*inv; rel_unit[i*3+1] = dy*inv; rel_unit[i*3+2] = dz*inv;
    const float sigma = 4.0f / (float)RBFN;         // 0.25
    const float den = 1.0f / (2.0f * sigma * sigma);
    const float step = 4.0f / (float)(RBFN - 1);
    for (int r = 0; r < RBFN; r++) {
        float t = dist - step * (float)r;
        rbf[i*RBFN + r] = __expf(-t*t*den);
    }
    nbm[i] = mask[j] ? 1.0f : 0.0f;
    rowmapV[i]            = j;
    rowmapV[nn*kk + i]    = nn + j;
    rowmapV[2*nn*kk + i]  = 2*nn + j;
}

// ---------------------------------------------------------------------------
// WMMA GEMM: C[M,Nd] (f32) = gather(A)[M,Kd] (f16) @ B[Kd,Nd] (f16)
//   optional rowmap gather on A rows, optional bias, optional accumulate.
// Block = 128 threads = 4 wave32s in a 2x2 arrangement over a 64x64 tile,
// each wave owns a 32x32 subtile = 2x2 v_wmma_f32_16x16x32_f16.
// Staging is vectorized: A row addresses (incl. gather + M guard) are hoisted
// out of the K loop; all global traffic is 16B per request.
// ---------------------------------------------------------------------------
#define BM 64
#define BN 64
#define BK 32

__global__ void __launch_bounds__(128)
gemm_wmma_kernel(const half_t* __restrict__ A, const half_t* __restrict__ B,
                 float* __restrict__ C, int M, int Kd, int Nd,
                 const int* __restrict__ rowmap, const float* __restrict__ bias,
                 int accumulate) {
    __shared__ __align__(16) half_t As[BM][BK];
    __shared__ __align__(16) half_t Bs[BN][BK];   // B tile transposed: Bs[n][k]
    int tid  = threadIdx.x;
    int wave = tid >> 5, lane = tid & 31;
    int wrow = (wave >> 1) * 32, wcol = (wave & 1) * 32;
    int m0 = blockIdx.y * BM, n0 = blockIdx.x * BN;
    int lr = lane & 15, lh = lane >> 4;
    v8f acc[2][2] = {};

    // ---- A staging map: 2 threads per tile row, 16 halves (32B) each ----
    int ar_row  = tid >> 1;               // 0..63
    int ar_half = (tid & 1) * 16;         // 0 or 16
    int gm = m0 + ar_row;
    bool mvalid = gm < M;
    size_t arow = 0;
    if (mvalid) arow = (size_t)(rowmap ? rowmap[gm] : gm);
    const half_t* aptr = A + arow * Kd + ar_half;

    // ---- B staging map: thread owns (k row, 16-wide n chunk) ----
    int bk_row = tid >> 2;                // 0..31
    int bn_ch  = (tid & 3) * 16;          // 0,16,32,48
    const half_t* bptr = B + (size_t)bk_row * Nd + n0 + bn_ch;

    for (int k0 = 0; k0 < Kd; k0 += BK) {
        if (k0 + BK < Kd)
            __builtin_prefetch(&B[(size_t)(k0 + BK) * Nd + n0], 0, 1);
        // stage A tile (vector loads; gather already resolved)
        {
            v8h lo = {}, hi = {};
            if (mvalid) {
                lo = *(const v8h*)(aptr + k0);
                hi = *(const v8h*)(aptr + k0 + 8);
            }
            *(v8h*)&As[ar_row][ar_half]     = lo;
            *(v8h*)&As[ar_row][ar_half + 8] = hi;
        }
        // stage B tile: contiguous 16B global reads, transpose via LDS scatter
        {
            v8h lo = *(const v8h*)(bptr + (size_t)k0 * Nd);
            v8h hi = *(const v8h*)(bptr + (size_t)k0 * Nd + 8);
            for (int j = 0; j < 8; j++) {
                Bs[bn_ch + j][bk_row]     = lo[j];
                Bs[bn_ch + 8 + j][bk_row] = hi[j];
            }
        }
        __syncthreads();

        v16h afrag[2], bfrag[2];
        // A fragment: lane<16 -> row M=lr, K {0..7,16..23}; lane>=16 -> K {8..15,24..31}
        for (int mi = 0; mi < 2; mi++) {
            const v8h lo = *(const v8h*)&As[wrow + mi*16 + lr][lh*8];
            const v8h hi = *(const v8h*)&As[wrow + mi*16 + lr][lh*8 + 16];
            for (int e = 0; e < 8; e++) { afrag[mi][e] = lo[e]; afrag[mi][e+8] = hi[e]; }
        }
        // B fragment: lane<16 -> col N=lr, K {0..7,16..23}; lane>=16 -> K {8..15,24..31}
        for (int nj = 0; nj < 2; nj++) {
            const v8h lo = *(const v8h*)&Bs[wcol + nj*16 + lr][lh*8];
            const v8h hi = *(const v8h*)&Bs[wcol + nj*16 + lr][lh*8 + 16];
            for (int e = 0; e < 8; e++) { bfrag[nj][e] = lo[e]; bfrag[nj][e+8] = hi[e]; }
        }
        for (int mi = 0; mi < 2; mi++)
            for (int nj = 0; nj < 2; nj++)
                acc[mi][nj] = __builtin_amdgcn_wmma_f32_16x16x32_f16(
                    false, afrag[mi], false, bfrag[nj], (short)0, acc[mi][nj], false, false);
        __syncthreads();
    }

    // write back: VGPR e <-> row e + 8*(lane>=16); lane&15 <-> col
    for (int mi = 0; mi < 2; mi++) {
        int rbase = m0 + wrow + mi*16 + lh*8;
        for (int nj = 0; nj < 2; nj++) {
            int col = n0 + wcol + nj*16 + lr;
            if (col >= Nd) continue;
            float bv = bias ? bias[col] : 0.0f;
            for (int e = 0; e < 8; e++) {
                int row = rbase + e;
                if (row < M) {
                    size_t o = (size_t)row * Nd + col;
                    float v = acc[mi][nj][e] + bv;
                    C[o] = accumulate ? (C[o] + v) : v;
                }
            }
        }
    }
}

// ---------------------------------------------------------------------------
// LayerNorm over C0=256 (blockDim must be 256); writes f16 for the GEMMs
// ---------------------------------------------------------------------------
__global__ void ln_kernel(const float* __restrict__ x, half_t* __restrict__ y, int C) {
    __shared__ float red[256];
    int n = blockIdx.x, c = threadIdx.x;
    float v = x[(size_t)n * C + c];
    red[c] = v; __syncthreads();
    for (int s = 128; s > 0; s >>= 1) { if (c < s) red[c] += red[c + s]; __syncthreads(); }
    float mean = red[0] / (float)C;
    __syncthreads();
    float d = v - mean;
    red[c] = d * d; __syncthreads();
    for (int s = 128; s > 0; s >>= 1) { if (c < s) red[c] += red[c + s]; __syncthreads(); }
    float var = red[0] / (float)C;
    y[(size_t)n * C + c] = (half_t)(d * rsqrtf(var + 1e-6f));
}

// ---------------------------------------------------------------------------
// km += rbf @ We ; ge += rbf @ Wer   (R=16 inner dim, VALU)
// ---------------------------------------------------------------------------
__global__ void rbf_add_kernel(float* __restrict__ km, float* __restrict__ ge,
                               const float* __restrict__ rbf,
                               const float* __restrict__ We, const float* __restrict__ Wer) {
    int i = blockIdx.x;          // neighbor pair index (NK)
    int h = threadIdx.x;         // 0..127
    float s0 = 0.0f, s1 = 0.0f;
    for (int r = 0; r < RBFN; r++) {
        float rv = rbf[(size_t)i * RBFN + r];
        s0 += rv * We[r * HDIM + h];
        s1 += rv * Wer[r * HDIM + h];
    }
    km[(size_t)i * HDIM + h] += s0;
    ge[(size_t)i * HDIM + h] += s1;
}

// ---------------------------------------------------------------------------
// Attention: one block per point, one wave per head (wave32)
// ---------------------------------------------------------------------------
__global__ void __launch_bounds__(128)
attn_kernel(const float* __restrict__ q, const float* __restrict__ km,
            const float* __restrict__ v0, const float* __restrict__ ge,
            const float* __restrict__ v1m, const float* __restrict__ relu,
            const float* __restrict__ nbm, float* __restrict__ o0, float* __restrict__ o1) {
    __shared__ float slog[HEADS][KNN];
    int n = blockIdx.x;
    int tid = threadIdx.x, h = tid >> 5, lane = tid & 31;
    if (lane < KNN) {
        int nk = n * KNN + lane;
        const float* qp = q  + (size_t)n  * HDIM + h * DH;
        const float* kp = km + (size_t)nk * HDIM + h * DH;
        float s = 0.0f;
        for (int d = 0; d < DH; d++) s += qp[d] * kp[d];
        s *= 0.17677669529663687f;                 // 1/sqrt(32)
        if (nbm[nk] == 0.0f) s = -1e9f;
        slog[h][lane] = s;
    }
    __syncthreads();
    float l[KNN]; float mx = -INFINITY;
    for (int t = 0; t < KNN; t++) { l[t] = slog[h][t]; mx = fmaxf(mx, l[t]); }
    float sum = 0.0f;
    for (int t = 0; t < KNN; t++) { l[t] = __expf(l[t] - mx); sum += l[t]; }
    float isum = 1.0f / sum;
    int d = lane;
    float a0 = 0.0f, ax = 0.0f, ay = 0.0f, az = 0.0f;
    const size_t NKs = (size_t)NPTS * KNN;
    for (int t = 0; t < KNN; t++) {
        int nk = n * KNN + t;
        float w = l[t] * isum;
        a0 += w * v0[(size_t)nk * HDIM + h * DH + d];
        float g  = ge[(size_t)nk * HDIM + h * DH + d];
        float rx = relu[nk*3+0], ry = relu[nk*3+1], rz = relu[nk*3+2];
        ax += w * (v1m[((size_t)0*NKs + nk) * HDIM + h*DH + d] + g * rx);
        ay += w * (v1m[((size_t)1*NKs + nk) * HDIM + h*DH + d] + g * ry);
        az += w * (v1m[((size_t)2*NKs + nk) * HDIM + h*DH + d] + g * rz);
    }
    o0[(size_t)n * HDIM + h*DH + d] = a0;
    o1[((size_t)0*NPTS + n) * HDIM + h*DH + d] = ax;
    o1[((size_t)1*NPTS + n) * HDIM + h*DH + d] = ay;
    o1[((size_t)2*NPTS + n) * HDIM + h*DH + d] = az;
}

// ---------------------------------------------------------------------------
// gelu(x + bias) -> f16 (tanh approximation, matches jax.nn.gelu default)
// ---------------------------------------------------------------------------
__global__ void gelu_bias_cvt_kernel(const float* __restrict__ x, const float* __restrict__ bias,
                                     half_t* __restrict__ y, int total, int cols) {
    int i = blockIdx.x * blockDim.x + threadIdx.x;
    if (i >= total) return;
    float v = x[i] + bias[i % cols];
    float t = 0.7978845608028654f * (v + 0.044715f * v * v * v);
    y[i] = (half_t)(0.5f * v * (1.0f + tanhf(t)));
}

// ---------------------------------------------------------------------------
// f1 += tmp * sigmoid(gateraw)   (f1 layout [3][N][C1])
// ---------------------------------------------------------------------------
__global__ void f1_gate_kernel(float* __restrict__ f1, const float* __restrict__ tmp,
                               const float* __restrict__ gateraw, int total) {
    int i = blockIdx.x * blockDim.x + threadIdx.x;
    if (i >= total) return;
    int rem = i % (NPTS * C1);
    int n = rem / C1, dd = rem % C1;
    float g = 1.0f / (1.0f + __expf(-gateraw[(size_t)n * C1 + dd]));
    f1[i] += tmp[i] * g;
}

// ---------------------------------------------------------------------------
// pred_vector[n][v] = sum_c f1[v][n][c] * wv[c]
// ---------------------------------------------------------------------------
__global__ void final_vec_kernel(const float* __restrict__ f1, const float* __restrict__ wv,
                                 float* __restrict__ out) {
    int n = blockIdx.x * blockDim.x + threadIdx.x;
    if (n >= NPTS) return;
    for (int v = 0; v < 3; v++) {
        float s = 0.0f;
        for (int c = 0; c < C1; c++) s += f1[((size_t)v * NPTS + n) * C1 + c] * wv[c];
        out[(size_t)n * 3 + v] = s;
    }
}

// ---------------------------------------------------------------------------
// Host orchestration
// ---------------------------------------------------------------------------
static inline char* carve(char*& p, size_t bytes) {
    char* r = p;
    p += (bytes + 255) & ~(size_t)255;
    return r;
}

extern "C" void kernel_launch(void* const* d_in, const int* in_sizes, int n_in,
                              void* d_out, int out_size, void* d_ws, size_t ws_size,
                              hipStream_t stream) {
    (void)in_sizes; (void)n_in; (void)out_size; (void)ws_size;
    const float* feats = (const float*)d_in[0];
    const float* coors = (const float*)d_in[1];
    const unsigned char* mask = (const unsigned char*)d_in[2];
    const float* Wemb = (const float*)d_in[3];
    const float* Wq   = (const float*)d_in[4];
    const float* Wk   = (const float*)d_in[5];
    const float* Wv0  = (const float*)d_in[6];
    const float* Wvr  = (const float*)d_in[7];
    const float* We   = (const float*)d_in[8];
    const float* Wer  = (const float*)d_in[9];
    const float* Wv1  = (const float*)d_in[10];
    const float* Wo0  = (const float*)d_in[11];
    const float* Wo1  = (const float*)d_in[12];
    const float* Wf1  = (const float*)d_in[13];
    const float* bf1  = (const float*)d_in[14];
    const float* Wf2  = (const float*)d_in[15];
    const float* bf2  = (const float*)d_in[16];
    const float* Wg   = (const float*)d_in[17];
    const float* Wvff = (const float*)d_in[18];
    const float* Ws   = (const float*)d_in[19];
    const float* bs   = (const float*)d_in[20];
    const float* wv   = (const float*)d_in[21];
    float* out = (float*)d_out;

    // -------- workspace carve (~650 MB) --------
    char* p = (char*)d_ws;
    int*    idx      = (int*)carve(p, (size_t)NK * 4);
    int*    rowmapV  = (int*)carve(p, (size_t)3 * NK * 4);
    float*  rel_unit = (float*)carve(p, (size_t)NK * 3 * 4);
    float*  rbf      = (float*)carve(p, (size_t)NK * RBFN * 4);
    float*  nbm      = (float*)carve(p, (size_t)NK * 4);
    float*  f0       = (float*)carve(p, (size_t)NPTS * C0 * 4);
    half_t* f0n16    = (half_t*)carve(p, (size_t)NPTS * C0 * 2);
    float*  f1s      = (float*)carve(p, (size_t)3 * NPTS * C1 * 4);
    half_t* f1h      = (half_t*)carve(p, (size_t)3 * NPTS * C1 * 2);
    half_t* featsH   = (half_t*)carve(p, (size_t)NPTS * DIN * 2);
    float*  qb       = (float*)carve(p, (size_t)NPTS * HDIM * 4);
    float*  km       = (float*)carve(p, (size_t)NK * HDIM * 4);
    float*  v0b      = (float*)carve(p, (size_t)NK * HDIM * 4);
    float*  geb      = (float*)carve(p, (size_t)NK * HDIM * 4);
    float*  v1m      = (float*)carve(p, (size_t)3 * NK * HDIM * 4);
    float*  o0       = (float*)carve(p, (size_t)NPTS * HDIM * 4);
    half_t* o0h      = (half_t*)carve(p, (size_t)NPTS * HDIM * 2);
    float*  o1       = (float*)carve(p, (size_t)3 * NPTS * HDIM * 4);
    half_t* o1h      = (half_t*)carve(p, (size_t)3 * NPTS * HDIM * 2);
    float*  h1       = (float*)carve(p, (size_t)NPTS * 4 * C0 * 4);
    half_t* h1h      = (half_t*)carve(p, (size_t)NPTS * 4 * C0 * 2);
    float*  gateraw  = (float*)carve(p, (size_t)NPTS * C1 * 4);
    float*  f1tmp    = (float*)carve(p, (size_t)3 * NPTS * C1 * 4);
    half_t* WembH = (half_t*)carve(p, (size_t)C0 * DIN * 2);
    half_t* WqH   = (half_t*)carve(p, (size_t)DEPTH * C0 * HDIM * 2);
    half_t* WkH   = (half_t*)carve(p, (size_t)DEPTH * C0 * HDIM * 2);
    half_t* Wv0H  = (half_t*)carve(p, (size_t)DEPTH * C0 * HDIM * 2);
    half_t* WvrH  = (half_t*)carve(p, (size_t)DEPTH * C0 * HDIM * 2);
    half_t* Wv1H  = (half_t*)carve(p, (size_t)DEPTH * C1 * HDIM * 2);
    half_t* Wo0H  = (half_t*)carve(p, (size_t)DEPTH * HDIM * C0 * 2);
    half_t* Wo1H  = (half_t*)carve(p, (size_t)DEPTH * HDIM * C1 * 2);
    half_t* Wf1H  = (half_t*)carve(p, (size_t)DEPTH * C0 * 4 * C0 * 2);
    half_t* Wf2H  = (half_t*)carve(p, (size_t)DEPTH * 4 * C0 * C0 * 2);
    half_t* WgH   = (half_t*)carve(p, (size_t)DEPTH * C0 * C1 * 2);
    half_t* WvffH = (half_t*)carve(p, (size_t)DEPTH * C1 * C1 * 2);
    half_t* WsH   = (half_t*)carve(p, (size_t)C0 * DOUT * 2);

    auto cvt = [&](const float* s, half_t* d, int n) {
        cvt_f16_kernel<<<(n + 255) / 256, 256, 0, stream>>>(s, d, n);
    };
    auto ggrid = [](int M, int Nd) { return dim3((unsigned)((Nd + 63) / 64), (unsigned)((M + 63) / 64)); };
    auto gemm = [&](const half_t* A, const half_t* B, float* C, int M, int Kd, int Nd,
                    const int* rmap, const float* bias, int acc) {
        gemm_wmma_kernel<<<ggrid(M, Nd), 128, 0, stream>>>(A, B, C, M, Kd, Nd, rmap, bias, acc);
    };

    // -------- weight + input conversions --------
    cvt(feats, featsH, NPTS * DIN);
    cvt(Wemb, WembH, C0 * DIN);
    cvt(Wq,  WqH,  DEPTH * C0 * HDIM);
    cvt(Wk,  WkH,  DEPTH * C0 * HDIM);
    cvt(Wv0, Wv0H, DEPTH * C0 * HDIM);
    cvt(Wvr, WvrH, DEPTH * C0 * HDIM);
    cvt(Wv1, Wv1H, DEPTH * C1 * HDIM);
    cvt(Wo0, Wo0H, DEPTH * HDIM * C0);
    cvt(Wo1, Wo1H, DEPTH * HDIM * C1);
    cvt(Wf1, Wf1H, DEPTH * C0 * 4 * C0);
    cvt(Wf2, Wf2H, DEPTH * 4 * C0 * C0);
    cvt(Wg,  WgH,  DEPTH * C0 * C1);
    cvt(Wvff, WvffH, DEPTH * C1 * C1);
    cvt(Ws, WsH, C0 * DOUT);

    // -------- kNN + geometry --------
    size_t knn_sh = ((size_t)NPTS + 256) * sizeof(float) + 256 * sizeof(int);
    knn_kernel<<<NPTS, 256, knn_sh, stream>>>(coors, mask, idx, NPTS, KNN);
    geom_kernel<<<(NK + 255) / 256, 256, 0, stream>>>(coors, mask, idx, rel_unit, rbf, nbm,
                                                      rowmapV, NPTS, KNN);

    (void)hipMemsetAsync(f1s, 0, (size_t)3 * NPTS * C1 * 4, stream);
    (void)hipMemsetAsync(f1h, 0, (size_t)3 * NPTS * C1 * 2, stream);

    // -------- embedding: f0 = feats @ Wemb --------
    gemm(featsH, WembH, f0, NPTS, DIN, C0, nullptr, nullptr, 0);

    for (int l = 0; l < DEPTH; l++) {
        const half_t* wq = WqH  + (size_t)l * C0 * HDIM;
        const half_t* wk = WkH  + (size_t)l * C0 * HDIM;
        const half_t* w0 = Wv0H + (size_t)l * C0 * HDIM;
        const half_t* wr = WvrH + (size_t)l * C0 * HDIM;
        const half_t* w1 = Wv1H + (size_t)l * C1 * HDIM;
        const half_t* wo0 = Wo0H + (size_t)l * HDIM * C0;
        const half_t* wo1 = Wo1H + (size_t)l * HDIM * C1;
        const half_t* wf1 = Wf1H + (size_t)l * C0 * 4 * C0;
        const half_t* wf2 = Wf2H + (size_t)l * 4 * C0 * C0;
        const half_t* wg  = WgH  + (size_t)l * C0 * C1;
        const half_t* wvf = WvffH + (size_t)l * C1 * C1;

        // LN -> f16 activations
        ln_kernel<<<NPTS, 256, 0, stream>>>(f0, f0n16, C0);
        // projections (gathered GEMMs route the f0[idx] expansion through WMMA)
        gemm(f0n16, wq, qb, NPTS, C0, HDIM, nullptr, nullptr, 0);
        gemm(f0n16, wk, km,  NK, C0, HDIM, idx, nullptr, 0);
        gemm(f0n16, w0, v0b, NK, C0, HDIM, idx, nullptr, 0);
        gemm(f0n16, wr, geb, NK, C0, HDIM, idx, nullptr, 0);
        gemm(f1h,   w1, v1m, 3 * NK, C1, HDIM, rowmapV, nullptr, 0);
        // RBF projections (R=16 inner dim, VALU)
        rbf_add_kernel<<<NK, HDIM, 0, stream>>>(km, geb, rbf,
                                                We + (size_t)l * RBFN * HDIM,
                                                Wer + (size_t)l * RBFN * HDIM);
        // softmax attention over K=16 neighbors
        attn_kernel<<<NPTS, 128, 0, stream>>>(qb, km, v0b, geb, v1m, rel_unit, nbm, o0, o1);
        cvt(o0, o0h, NPTS * HDIM);
        cvt(o1, o1h, 3 * NPTS * HDIM);
        // residual projections
        gemm(o0h, wo0, f0, NPTS, HDIM, C0, nullptr, nullptr, 1);
        gemm(o1h, wo1, f1s, 3 * NPTS, HDIM, C1, nullptr, nullptr, 1);
        cvt(f1s, f1h, 3 * NPTS * C1);
        // feed-forward + invariant gating
        ln_kernel<<<NPTS, 256, 0, stream>>>(f0, f0n16, C0);
        gemm(f0n16, wf1, h1, NPTS, C0, 4 * C0, nullptr, nullptr, 0);
        gelu_bias_cvt_kernel<<<(NPTS * 4 * C0 + 255) / 256, 256, 0, stream>>>(
            h1, bf1 + (size_t)l * 4 * C0, h1h, NPTS * 4 * C0, 4 * C0);
        gemm(h1h, wf2, f0, NPTS, 4 * C0, C0, nullptr, bf2 + (size_t)l * C0, 1);
        gemm(f0n16, wg, gateraw, NPTS, C0, C1, nullptr, nullptr, 0);
        gemm(f1h, wvf, f1tmp, 3 * NPTS, C1, C1, nullptr, nullptr, 0);
        f1_gate_kernel<<<(3 * NPTS * C1 + 255) / 256, 256, 0, stream>>>(
            f1s, f1tmp, gateraw, 3 * NPTS * C1);
        cvt(f1s, f1h, 3 * NPTS * C1);
    }

    // -------- readout --------
    // pred_scalar = f0 @ Ws + bs  -> out[0 : N*DOUT]  (refresh f16 copy of f0 first)
    cvt(f0, f0n16, NPTS * C0);
    gemm(f0n16, WsH, out, NPTS, C0, DOUT, nullptr, bs, 0);
    // pred_vector -> out[N*DOUT : N*DOUT + N*3]
    final_vec_kernel<<<(NPTS + 255) / 256, 256, 0, stream>>>(f1s, wv, out + (size_t)NPTS * DOUT);
}
